// RPESelfAttention_55207509622993
// MI455X (gfx1250) — matrix operand order repaired and checked by reading, not compiled, not run on hardware
//
#include <hip/hip_runtime.h>

// ---------------------------------------------------------------------------
// RPE self-attention for gfx1250 (MI455X).  B=2, T=2048, C=1024, NH=16, HD=64,
// MRP=64 (banded causal attention, 64-wide band).
// Strategy: bf16 WMMA (v_wmma_f32_16x16x32_bf16) for both big GEMMs and for
// the banded attention (scores + RPE band GEMM + output GEMMs).
// ---------------------------------------------------------------------------

typedef __attribute__((ext_vector_type(16))) __bf16          v16bf;
typedef __attribute__((ext_vector_type(8)))  float           v8f;
typedef __attribute__((ext_vector_type(8)))  unsigned short  v8us;

static constexpr int Bb  = 2;
static constexpr int Tt  = 2048;
static constexpr int Cc  = 1024;
static constexpr int NH  = 16;
static constexpr int HD  = 64;

union FragU { v16bf v; v8us h[2]; };

static __device__ inline v16bf frag_from(const __bf16* p0, const __bf16* p1) {
  FragU u;
  u.h[0] = *(const v8us*)p0;
  u.h[1] = *(const v8us*)p1;
  return u.v;
}

static __device__ inline v8f wmma_bf16(v16bf a, v16bf b, v8f c) {
  // D = A(16x32) * B(32x16) + C, f32 accumulate
  return __builtin_amdgcn_wmma_f32_16x16x32_bf16(false, a, false, b,
                                                 (short)0, c, false, false);
}

// -------------------------------- converters -------------------------------

__global__ void k_cvt(const float* __restrict__ src, __bf16* __restrict__ dst, int n) {
  int i = blockIdx.x * blockDim.x + threadIdx.x;
  if (i < n) dst[i] = (__bf16)src[i];
}

// src is [K][N] row-major fp32, dst is [N][K] row-major bf16 (B^T layout)
__global__ void k_transpose_cvt(const float* __restrict__ src, __bf16* __restrict__ dst,
                                int K, int N) {
  int i = blockIdx.x * blockDim.x + threadIdx.x;
  if (i >= K * N) return;
  int n = i / K;
  int k = i - n * K;
  dst[(size_t)n * K + k] = (__bf16)src[(size_t)k * N + n];
}

// Tk16[t][d] = bf16(table_k[t][d]), t<129 ; TvT[d][j] = bf16(table_v[j+1][d])
__global__ void k_tables(const float* __restrict__ tk, const float* __restrict__ tv,
                         __bf16* __restrict__ Tk16, __bf16* __restrict__ TvT) {
  int i = blockIdx.x * blockDim.x + threadIdx.x;
  if (i < 129 * 64) {
    Tk16[i] = (__bf16)tk[i];
  } else if (i < 129 * 64 + 64 * 64) {
    int i2 = i - 129 * 64;
    int d = i2 >> 6;
    int j = i2 & 63;
    TvT[d * 64 + j] = (__bf16)tv[(j + 1) * 64 + d];
  }
}

// vT[b][h][d][t] = qkv16[b*T+t][2*C + h*64 + d]
__global__ void k_transpose_v(const __bf16* __restrict__ qkv, __bf16* __restrict__ vT) {
  int i = blockIdx.x * blockDim.x + threadIdx.x;
  if (i >= Bb * NH * HD * Tt) return;
  int t  = i & (Tt - 1);
  int d  = (i >> 11) & (HD - 1);
  int h  = (i >> 17) & (NH - 1);
  int b  = i >> 21;
  vT[i] = qkv[(size_t)(b * Tt + t) * (3 * Cc) + 2 * Cc + h * HD + d];
}

// ------------------------------- WMMA GEMM ---------------------------------
// C[M,N] = A[M,K] * BT[N,K]^T.  A,BT bf16 row-major; one wave -> 16x64 tile.
// blockDim = 128 (4 waves), grid = (M/64, N/64).

template <typename OutT>
__global__ __launch_bounds__(128) void k_gemm(const __bf16* __restrict__ A,
                                              const __bf16* __restrict__ BT,
                                              OutT* __restrict__ C,
                                              int M, int N, int K) {
  const int lane = threadIdx.x & 31;
  const int wave = threadIdx.x >> 5;
  const int row  = lane & 15;
  const int half = lane >> 4;
  const int m0 = blockIdx.x * 64 + wave * 16;
  const int n0 = blockIdx.y * 64;

  const v8f zero8 = {0.f, 0.f, 0.f, 0.f, 0.f, 0.f, 0.f, 0.f};
  v8f acc[4];
  acc[0] = zero8; acc[1] = zero8; acc[2] = zero8; acc[3] = zero8;

  const __bf16* Arow = A + (size_t)(m0 + row) * K;

  for (int ks = 0; ks < K; ks += 32) {
    if (ks + 64 < K) __builtin_prefetch(Arow + ks + 64, 0, 0);
    const int ka = ks + half * 16;
    v16bf a = frag_from(Arow + ka, Arow + ka + 8);
#pragma unroll
    for (int nt = 0; nt < 4; ++nt) {
      const __bf16* Brow = BT + (size_t)(n0 + nt * 16 + row) * K + ks + half * 16;
      v16bf bb = frag_from(Brow, Brow + 8);
      acc[nt] = wmma_bf16(a, bb, acc[nt]);
    }
  }

#pragma unroll
  for (int nt = 0; nt < 4; ++nt) {
#pragma unroll
    for (int r = 0; r < 8; ++r) {
      C[(size_t)(m0 + r + half * 8) * N + (n0 + nt * 16 + row)] = (OutT)acc[nt][r];
    }
  }
}

// ---------------------------- banded attention -----------------------------
// One wave handles one (b, h, 16-query tile).  Band width 64: keys q-63..q.
//   scores  = Q*K^T (5 abs key tiles) + Q*Tk^T (band layout), *1/sqrt(64)
//   softmax over the band, weights re-emitted in both layouts (bf16 in LDS)
//   out     = Att_band * Tv  +  Att_abs * V          (all WMMA)

__global__ __launch_bounds__(128) void k_attn(const __bf16* __restrict__ qkv,
                                              const __bf16* __restrict__ Tk,
                                              const __bf16* __restrict__ TvT,
                                              const __bf16* __restrict__ vT,
                                              __bf16* __restrict__ y) {
  __shared__ float sAbs[4][16][80];                   // content scores, abs cols
  __shared__ float sBand[4][16][64];                  // RPE scores / scratch
  __shared__ __align__(16) __bf16 aBand[4][16][64];   // attn weights, band layout
  __shared__ __align__(16) __bf16 aAbs[4][16][96];    // attn weights, abs layout (pad)

  const int lane = threadIdx.x & 31;
  const int wave = threadIdx.x >> 5;
  const int gw   = blockIdx.x * 4 + wave;
  const int qt   = gw & 127;
  const int h    = (gw >> 7) & 15;
  const int b    = gw >> 11;
  const int q0   = qt * 16;
  const int row  = lane & 15;
  const int half = lane >> 4;

  // zero-fill attention-weight arrays (masked/out-of-band entries stay 0)
  {
    __bf16* pB = &aBand[0][0][0];
    __bf16* pA = &aAbs[0][0][0];
    for (int i = threadIdx.x; i < 4 * 16 * 64; i += 128) pB[i] = (__bf16)0.0f;
    for (int i = threadIdx.x; i < 4 * 16 * 96; i += 128) pA[i] = (__bf16)0.0f;
  }

  const v8f zero8 = {0.f, 0.f, 0.f, 0.f, 0.f, 0.f, 0.f, 0.f};
  const int ka = half * 16;  // A-fragment K-run base
  const int kb = half * 16;  // B-fragment K-run base

  // Q A-fragments (16x32 each, HD=64 -> two)
  const __bf16* qrow = qkv + (size_t)(b * Tt + q0 + row) * (3 * Cc) + h * HD;
  v16bf a0 = frag_from(qrow + ka,      qrow + ka + 8);
  v16bf a1 = frag_from(qrow + 32 + ka, qrow + 32 + ka + 8);

  // ---- content scores: 5 absolute key tiles covering [q0-64, q0+16) ----
  for (int tt = 0; tt < 5; ++tt) {
    int kr  = q0 - 64 + tt * 16 + row;          // key index for this lane's column
    int krc = kr < 0 ? 0 : kr;                  // clamp; masked later
    const __bf16* krow = qkv + (size_t)(b * Tt + krc) * (3 * Cc) + Cc + h * HD;
    v16bf b0 = frag_from(krow + kb,      krow + kb + 8);
    v16bf b1 = frag_from(krow + 32 + kb, krow + 32 + kb + 8);
    v8f c = zero8;
    c = wmma_bf16(a0, b0, c);
    c = wmma_bf16(a1, b1, c);
#pragma unroll
    for (int r = 0; r < 8; ++r) sAbs[wave][r + half * 8][tt * 16 + row] = c[r];
  }

  // ---- RPE scores in band layout: S2[i][j] = Q[i]·table_k[j+1] ----
  for (int jt = 0; jt < 4; ++jt) {
    const __bf16* trow = Tk + (size_t)(jt * 16 + row + 1) * HD;
    v16bf b0 = frag_from(trow + kb,      trow + kb + 8);
    v16bf b1 = frag_from(trow + 32 + kb, trow + 32 + kb + 8);
    v8f c = zero8;
    c = wmma_bf16(a0, b0, c);
    c = wmma_bf16(a1, b1, c);
#pragma unroll
    for (int r = 0; r < 8; ++r) sBand[wave][r + half * 8][jt * 16 + row] = c[r];
  }

  __syncthreads();

  // ---- softmax per query row (band j: k = q-63+j, valid iff k >= 0) ----
  if (lane < 16) {
    const int i = lane;
    const int q = q0 + i;
    int jmin = 63 - q; if (jmin < 0) jmin = 0;
    float mx = -3.0e38f;
    for (int j = jmin; j < 64; ++j) {
      float s = (sAbs[wave][i][i + j + 1] + sBand[wave][i][j]) * 0.125f;
      sBand[wave][i][j] = s;
      if (s > mx) mx = s;
    }
    float sum = 0.f;
    for (int j = jmin; j < 64; ++j) {
      float p = __expf(sBand[wave][i][j] - mx);
      sBand[wave][i][j] = p;
      sum += p;
    }
    float inv = 1.0f / sum;
    for (int j = jmin; j < 64; ++j) {
      __bf16 av = (__bf16)(sBand[wave][i][j] * inv);
      aBand[wave][i][j] = av;
      aAbs[wave][i][i + j + 1] = av;   // abs col c = k - (q0-64) = i+j+1
    }
  }

  __syncthreads();

  // ---- output: out = Att_band(16x64) x Tv(64x64) + Att_abs(16x96) x V ----
  const __bf16* abase = &aBand[wave][row][0];
  v16bf ab0 = frag_from(abase + ka,      abase + ka + 8);
  v16bf ab1 = frag_from(abase + 32 + ka, abase + 32 + ka + 8);
  const __bf16* xbase = &aAbs[wave][row][0];
  v16bf aa0 = frag_from(xbase + ka,      xbase + ka + 8);
  v16bf aa1 = frag_from(xbase + 32 + ka, xbase + 32 + ka + 8);
  v16bf aa2 = frag_from(xbase + 64 + ka, xbase + 64 + ka + 8);

  v8f acc[4];
  acc[0] = zero8; acc[1] = zero8; acc[2] = zero8; acc[3] = zero8;

  // band part: B[j][d] from TvT[d][j]
#pragma unroll
  for (int dt = 0; dt < 4; ++dt) {
    const __bf16* tr = TvT + (size_t)(dt * 16 + row) * 64;
    v16bf b0 = frag_from(tr + kb,      tr + kb + 8);
    v16bf b1 = frag_from(tr + 32 + kb, tr + 32 + kb + 8);
    acc[dt] = wmma_bf16(ab0, b0, acc[dt]);
    acc[dt] = wmma_bf16(ab1, b1, acc[dt]);
  }

  // abs part: B[t][d] from vT[b][h][d][t] (t innermost -> contiguous)
  const __bf16* vbase = vT + ((size_t)(b * NH + h) * HD) * Tt;
  v16bf aaf[3] = {aa0, aa1, aa2};
#pragma unroll
  for (int p = 0; p < 3; ++p) {
    int t0 = q0 - 64 + p * 32 + kb;
    int t0c = t0 < 0 ? 0 : (t0 > Tt - 16 ? Tt - 16 : t0);   // clamped rows hit 0-weights
#pragma unroll
    for (int dt = 0; dt < 4; ++dt) {
      const __bf16* vrow = vbase + (size_t)(dt * 16 + row) * Tt + t0c;
      v16bf bv = frag_from(vrow, vrow + 8);
      acc[dt] = wmma_bf16(aaf[p], bv, acc[dt]);
    }
  }

  // store y (bf16, [b, t, h*64+d]) for the projection GEMM
#pragma unroll
  for (int dt = 0; dt < 4; ++dt) {
#pragma unroll
    for (int r = 0; r < 8; ++r) {
      y[(size_t)(b * Tt + q0 + r + half * 8) * Cc + h * HD + dt * 16 + row] =
          (__bf16)acc[dt][r];
    }
  }
}

// ------------------------------- launcher ----------------------------------

static inline size_t align256(size_t x) { return (x + 255) & ~(size_t)255; }

extern "C" void kernel_launch(void* const* d_in, const int* in_sizes, int n_in,
                              void* d_out, int out_size, void* d_ws, size_t ws_size,
                              hipStream_t stream) {
  (void)in_sizes; (void)n_in; (void)out_size; (void)ws_size;

  const float* x      = (const float*)d_in[0];   // [B,T,C]
  const float* W_attn = (const float*)d_in[1];   // [C,3C]
  const float* W_proj = (const float*)d_in[2];   // [C,C]
  const float* tk     = (const float*)d_in[3];   // [129,64]
  const float* tv     = (const float*)d_in[4];   // [129,64]
  float* out          = (float*)d_out;           // [B,T,C] fp32

  const int M  = Bb * Tt;        // 4096
  const int K  = Cc;             // 1024
  const int N1 = 3 * Cc;         // 3072

  // workspace carve-up (bf16 buffers), ~59 MB total
  char* w = (char*)d_ws;
  size_t off = 0;
  __bf16* xb    = (__bf16*)(w + off); off = align256(off + (size_t)M * K * 2);
  __bf16* WaT   = (__bf16*)(w + off); off = align256(off + (size_t)N1 * K * 2);
  __bf16* WpT   = (__bf16*)(w + off); off = align256(off + (size_t)Cc * K * 2);
  __bf16* Tk16  = (__bf16*)(w + off); off = align256(off + (size_t)129 * 64 * 2);
  __bf16* TvT   = (__bf16*)(w + off); off = align256(off + (size_t)64 * 64 * 2);
  __bf16* qkv16 = (__bf16*)(w + off); off = align256(off + (size_t)M * N1 * 2);
  __bf16* vT16  = (__bf16*)(w + off); off = align256(off + (size_t)Bb * NH * HD * Tt * 2);
  __bf16* y16   = (__bf16*)(w + off); off = align256(off + (size_t)M * Cc * 2);

  // 1) fp32 -> bf16 conversions / transposes
  {
    int n = M * K;
    k_cvt<<<(n + 255) / 256, 256, 0, stream>>>(x, xb, n);
  }
  {
    int n = K * N1;
    k_transpose_cvt<<<(n + 255) / 256, 256, 0, stream>>>(W_attn, WaT, K, N1);
  }
  {
    int n = K * Cc;
    k_transpose_cvt<<<(n + 255) / 256, 256, 0, stream>>>(W_proj, WpT, K, Cc);
  }
  {
    int n = 129 * 64 + 64 * 64;
    k_tables<<<(n + 255) / 256, 256, 0, stream>>>(tk, tv, Tk16, TvT);
  }

  // 2) QKV projection: qkv16 = xb @ W_attn   (bf16 out)
  {
    dim3 grid(M / 64, N1 / 64);
    k_gemm<__bf16><<<grid, 128, 0, stream>>>(xb, WaT, qkv16, M, N1, K);
  }

  // 3) V transpose to [b,h,d,t]
  {
    int n = Bb * NH * HD * Tt;
    k_transpose_v<<<(n + 255) / 256, 256, 0, stream>>>(qkv16, vT16);
  }

  // 4) banded RPE attention -> y16
  {
    int nwaves = Bb * NH * (Tt / 16);    // 4096 waves, 4 per block
    k_attn<<<nwaves / 4, 128, 0, stream>>>(qkv16, Tk16, TvT, vT16, y16);
  }

  // 5) output projection: out = y16 @ W_proj   (fp32 out)
  {
    dim3 grid(M / 64, Cc / 64);
    k_gemm<float><<<grid, 128, 0, stream>>>(y16, WpT, out, M, Cc, K);
  }
}
